// UNetSwin_56934086476157
// MI455X (gfx1250) — matrix operand order repaired and checked by reading, not compiled, not run on hardware
//
#include <hip/hip_runtime.h>
#include <math.h>

// ---------------------------------------------------------------------------
// SwinV2 block for MI455X (gfx1250): wave32, WMMA f32_16x16x32_f16, big-LDS
// full-window fusion. All GEMMs via v_wmma; softmax/LN/l2norm in fp32 VALU.
// ---------------------------------------------------------------------------

typedef __attribute__((ext_vector_type(16))) _Float16 v16h;
typedef __attribute__((ext_vector_type(8)))  _Float16 v8h;
typedef __attribute__((ext_vector_type(4)))  _Float16 v4h;
typedef __attribute__((ext_vector_type(8)))  float    v8f;

#define H_    56
#define W_    56
#define WS_   7
#define SHIFT_ 3
#define NH    6
#define DIM   192
#define HD    32
#define NTOK  49
#define NPAD  64
#define NWIN  64
#define BATCH 16
#define LTOK  (H_*W_)          // 3136
#define TOKS  (BATCH*LTOK)     // 50176
#define FFN   768

// workspace layout (bytes)
#define WOFF_QKVH  0u          // 576x192 f16      = 221184 B
#define WOFF_PROJH 221184u     // 192x192 f16      =  73728 B
#define WOFF_FC1H  294912u     // 768x192 f16      = 294912 B
#define WOFF_FC2H  589824u     // 192x768 f16      = 294912 B
#define WOFF_RPB   884736u     // 6x49x49 f32      =  57624 B
#define WOFF_X1    950272u     // 50176x192 f32    = 38535168 B  (~37.6 MB total)

// ---- WMMA fragment helpers (layouts per CDNA5 ISA 7.12.2) ------------------

// A-matrix 16x32 f16: lane m=lane&15, kh=lane>>4; halfs[0..7]=A[m][k0+kh*8+0..7],
// halfs[8..15]=A[m][k0+16+kh*8+0..7]
static __device__ __forceinline__ v16h ldA(const _Float16* p, int ld, int m0, int k0, int lane) {
    int lm = lane & 15, kh = lane >> 4;
    const _Float16* r = p + (size_t)(m0 + lm) * ld + k0 + kh * 8;
    v8h lo = *(const v8h*)r;
    v8h hi = *(const v8h*)(r + 16);
    return __builtin_shufflevector(lo, hi, 0,1,2,3,4,5,6,7,8,9,10,11,12,13,14,15);
}

// B-matrix 32x16 f16 where B[k][n] = W[n0+n][k0+k] (W row-major NxK, i.e. X@W^T):
// lane n=lane&15, kh=lane>>4; halfs[0..7] = W[n0+n][k0+kh*8+0..7], halfs[8..15] = W[n0+n][k0+16+kh*8+0..7]
static __device__ __forceinline__ v16h ldBT(const _Float16* p, int ld, int n0, int k0, int lane) {
    int ln = lane & 15, kh = lane >> 4;
    const _Float16* r = p + (size_t)(n0 + ln) * ld + k0 + kh * 8;
    v8h lo = *(const v8h*)r;
    v8h hi = *(const v8h*)(r + 16);
    return __builtin_shufflevector(lo, hi, 0,1,2,3,4,5,6,7,8,9,10,11,12,13,14,15);
}

static __device__ __forceinline__ v8f wmma_f16(v16h a, v16h b, v8f c) {
    return __builtin_amdgcn_wmma_f32_16x16x32_f16(false, a, false, b, (short)0, c, false, false);
}

// C/D 16x16 f32: lane n=lane&15, mh=lane>>4; vgpr r -> M = mh*8+r
static __device__ __forceinline__ void stC(float* p, int ld, int m0, int n0, int lane, v8f c) {
    int ln = lane & 15, mh = lane >> 4;
#pragma unroll
    for (int r = 0; r < 8; ++r)
        p[(size_t)(m0 + mh * 8 + r) * ld + n0 + ln] = c[r];
}

// shift-mask region id of token t in window (wr,wc) (shifted coordinate space)
static __device__ __forceinline__ int regid(int wr, int wc, int t) {
    int sr = wr * WS_ + t / WS_;
    int sc = wc * WS_ + t % WS_;
    int rz = (sr < H_ - WS_) ? 0 : ((sr < H_ - SHIFT_) ? 1 : 2);
    int cz = (sc < W_ - WS_) ? 0 : ((sc < W_ - SHIFT_) ? 1 : 2);
    return rz * 3 + cz;
}

// ---- prep kernels ----------------------------------------------------------

__global__ __launch_bounds__(256) void prep_weights_kernel(
    const float* __restrict__ wqkv, const float* __restrict__ wproj,
    const float* __restrict__ fc1w, const float* __restrict__ fc2w,
    _Float16* __restrict__ dst)
{
    const int S0 = 576 * 192, S1 = 192 * 192, S2 = 768 * 192, S3 = 192 * 768;
    const int total = S0 + S1 + S2 + S3;
    for (int i = blockIdx.x * 256 + threadIdx.x; i < total; i += gridDim.x * 256) {
        float v;
        if      (i < S0)           v = wqkv[i];
        else if (i < S0 + S1)      v = wproj[i - S0];
        else if (i < S0 + S1 + S2) v = fc1w[i - S0 - S1];
        else                       v = fc2w[i - S0 - S1 - S2];
        dst[i] = (_Float16)v;
    }
}

// CPB-MLP: log-spaced table -> 512 relu -> NH, then 16*sigmoid gathered by rel idx
__global__ __launch_bounds__(256) void prep_rpb_kernel(
    const float* __restrict__ w1, const float* __restrict__ b1,
    const float* __restrict__ w2, float* __restrict__ rpb)
{
    __shared__ float btab[169 * NH];
    const int tid = threadIdx.x;
    if (tid < 169) {
        int ti = tid / 13, tj = tid % 13;
        float r0 = (float)(ti - 6) * (8.0f / 6.0f);
        float r1 = (float)(tj - 6) * (8.0f / 6.0f);
        float t0 = (r0 == 0.f ? 0.f : (r0 > 0.f ? 1.f : -1.f)) * (log2f(fabsf(r0) + 1.f) / 3.f);
        float t1 = (r1 == 0.f ? 0.f : (r1 > 0.f ? 1.f : -1.f)) * (log2f(fabsf(r1) + 1.f) / 3.f);
        float acc[NH];
#pragma unroll
        for (int h = 0; h < NH; ++h) acc[h] = 0.f;
        for (int j = 0; j < 512; ++j) {
            float hv = fmaxf(t0 * w1[j * 2 + 0] + t1 * w1[j * 2 + 1] + b1[j], 0.f);
#pragma unroll
            for (int h = 0; h < NH; ++h) acc[h] += hv * w2[h * 512 + j];
        }
#pragma unroll
        for (int h = 0; h < NH; ++h) btab[tid * NH + h] = acc[h];
    }
    __syncthreads();
    for (int e = tid; e < NH * NTOK * NTOK; e += 256) {
        int h = e / (NTOK * NTOK), rem = e % (NTOK * NTOK);
        int i = rem / NTOK, j = rem % NTOK;
        int dy = i / WS_ - j / WS_ + (WS_ - 1);
        int dx = i % WS_ - j % WS_ + (WS_ - 1);
        float v = btab[(dy * 13 + dx) * NH + h];
        rpb[e] = 16.f / (1.f + expf(-v));
    }
}

// ---- fused shifted-window attention (one window per workgroup) -------------
// dynamic LDS carve (bytes):
//   xh    [0      ,  24576)  64x192 f16  window input
//   tmpf  [24576  ,  75776)  64x200 f32  qkv staging / proj result (reused)
//   qn    [75776  , 100352)  64x192 f16  l2norm(q)*scale
//   kn    [100352 , 124928)  64x192 f16  l2norm(k)
//   vT    [124928 , 149504) 192x64  f16  v transposed (dim-major)
//   attnL [149504 , 166912)  64x68  f32  logits
//   attnH [166912 , 176128)  64x72  f16  softmax probs
//   outh  [176128 , 200704)  64x192 f16  attention output (pre-proj)
//   stat  [200704 , 201216)  mean/rstd
#define SMEM_ATTN 201216

__global__ __launch_bounds__(256) void swin_attn_kernel(
    const float* __restrict__ x, const float* __restrict__ q_bias,
    const float* __restrict__ v_bias, const float* __restrict__ logit_scale,
    const float* __restrict__ b_proj, const float* __restrict__ ln1_g,
    const float* __restrict__ ln1_b, const _Float16* __restrict__ wqkv_h,
    const _Float16* __restrict__ wproj_h, const float* __restrict__ rpb,
    float* __restrict__ x1)
{
    extern __shared__ float4 smemv[];
    char* smem = (char*)smemv;
    _Float16* xh    = (_Float16*)(smem);
    float*    tmpf  = (float*)   (smem + 24576);
    _Float16* qn    = (_Float16*)(smem + 75776);
    _Float16* kn    = (_Float16*)(smem + 100352);
    _Float16* vT    = (_Float16*)(smem + 124928);
    float*    attnL = (float*)   (smem + 149504);
    _Float16* attnH = (_Float16*)(smem + 166912);
    _Float16* outh  = (_Float16*)(smem + 176128);
    float*    stat  = (float*)   (smem + 200704);

    const int tid  = threadIdx.x;
    const int lane = tid & 31;
    const int wave = tid >> 5;
    const int b    = blockIdx.x >> 6;   // batch
    const int w    = blockIdx.x & 63;   // window
    const int wr   = w >> 3, wc = w & 7;

    // ---- stage 1: gather cyclically-shifted window, f32 -> f16 -------------
    for (int e = tid; e < NPAD * 48; e += 256) {
        int m = e / 48, g = e % 48;
        v4h hv;
        hv[0] = (_Float16)0.f; hv[1] = (_Float16)0.f; hv[2] = (_Float16)0.f; hv[3] = (_Float16)0.f;
        if (m < NTOK) {
            int sr = wr * WS_ + m / WS_, sc = wc * WS_ + m % WS_;
            int r = sr + SHIFT_; if (r >= H_) r -= H_;
            int c = sc + SHIFT_; if (c >= W_) c -= W_;
            const float4 v4 = *(const float4*)(x + ((size_t)b * LTOK + r * W_ + c) * DIM + g * 4);
            hv[0] = (_Float16)v4.x; hv[1] = (_Float16)v4.y;
            hv[2] = (_Float16)v4.z; hv[3] = (_Float16)v4.w;
        }
        *(v4h*)(xh + (size_t)m * DIM + g * 4) = hv;
    }
    __syncthreads();

    // ---- stage 2/3: QKV GEMM (3 passes: q,k,v) + normalize to f16 ----------
    for (int p = 0; p < 3; ++p) {
        // (64x192) @ (192->192)^T  : 4x12 tiles, 6 per wave, K=6 wmma steps
        for (int t = wave * 6; t < wave * 6 + 6; ++t) {
            int mt = t / 12, nt = t % 12;
            int col = nt * 16 + (lane & 15);
            float bv = (p == 0) ? q_bias[col] : ((p == 2) ? v_bias[col] : 0.f);
            v8f c;
#pragma unroll
            for (int r = 0; r < 8; ++r) c[r] = bv;
#pragma unroll
            for (int kk = 0; kk < 6; ++kk) {
                v16h a  = ldA (xh, DIM, mt * 16, kk * 32, lane);
                v16h bb = ldBT(wqkv_h, DIM, p * DIM + nt * 16, kk * 32, lane);
                c = wmma_f16(a, bb, c);
            }
            stC(tmpf, 200, mt * 16, nt * 16, lane, c);
        }
        __syncthreads();
        // per-(row,head) epilogue in fp32
        for (int e = tid; e < NPAD * NH; e += 256) {
            int row = e / NH, h = e % NH;
            const float* src = tmpf + (size_t)row * 200 + h * HD;
            if (p == 2) { // v: transpose to dim-major f16, zero pad rows
                for (int d = 0; d < HD; ++d)
                    vT[(size_t)(h * HD + d) * NPAD + row] =
                        (row < NTOK) ? (_Float16)src[d] : (_Float16)0.f;
            } else {
                _Float16* dst = (p == 0 ? qn : kn) + (size_t)row * DIM + h * HD;
                if (row < NTOK) {
                    float ss = 0.f;
                    for (int d = 0; d < HD; ++d) ss += src[d] * src[d];
                    float inv = 1.f / fmaxf(sqrtf(ss), 1e-12f);
                    if (p == 0)
                        inv *= expf(fminf(logit_scale[h], 4.6051702f)); // log(100)
                    for (int d = 0; d < HD; ++d) dst[d] = (_Float16)(src[d] * inv);
                } else {
                    for (int d = 0; d < HD; ++d) dst[d] = (_Float16)0.f;
                }
            }
        }
        __syncthreads();
    }

    // ---- stage 4/5: per-head attention -------------------------------------
    for (int h = 0; h < NH; ++h) {
        // logits = (qn*scale) @ kn^T : 4x4 tiles, 2 per wave, single K=32 step
        for (int t = wave * 2; t < wave * 2 + 2; ++t) {
            int mt = t >> 2, nt = t & 3;
            v8f c;
#pragma unroll
            for (int r = 0; r < 8; ++r) c[r] = 0.f;
            v16h a  = ldA (qn, DIM, mt * 16, h * HD, lane);
            v16h bb = ldBT(kn, DIM, nt * 16, h * HD, lane);
            c = wmma_f16(a, bb, c);
            int ln = lane & 15, mh = lane >> 4;
            int j = nt * 16 + ln;
#pragma unroll
            for (int r = 0; r < 8; ++r) {
                int i = mt * 16 + mh * 8 + r;
                float v;
                if (j < NTOK) {
                    float mk = (i < NTOK && regid(wr, wc, i) != regid(wr, wc, j)) ? -100.f : 0.f;
                    float rb = (i < NTOK) ? rpb[h * NTOK * NTOK + i * NTOK + j] : 0.f;
                    v = c[r] + rb + mk;
                } else {
                    v = -1e9f; // padded keys excluded
                }
                attnL[i * 68 + j] = v;
            }
        }
        __syncthreads();
        // softmax per row (fp32)
        if (tid < NPAD) {
            int i = tid;
            if (i < NTOK) {
                float mx = -1e30f;
                for (int j = 0; j < NTOK; ++j) mx = fmaxf(mx, attnL[i * 68 + j]);
                float s = 0.f;
                for (int j = 0; j < NTOK; ++j) {
                    float e = expf(attnL[i * 68 + j] - mx);
                    attnL[i * 68 + j] = e; s += e;
                }
                float is = 1.f / s;
                for (int j = 0; j < NTOK; ++j) attnH[i * 72 + j] = (_Float16)(attnL[i * 68 + j] * is);
                for (int j = NTOK; j < NPAD; ++j) attnH[i * 72 + j] = (_Float16)0.f;
            } else {
                for (int j = 0; j < NPAD; ++j) attnH[i * 72 + j] = (_Float16)0.f;
            }
        }
        __syncthreads();
        // out_h = attn @ v : 4x2 tiles, one per wave, K=2 steps
        {
            int mt = wave >> 1, nt = wave & 1;
            v8f c;
#pragma unroll
            for (int r = 0; r < 8; ++r) c[r] = 0.f;
#pragma unroll
            for (int kk = 0; kk < 2; ++kk) {
                v16h a  = ldA (attnH, 72, mt * 16, kk * 32, lane);
                v16h bb = ldBT(vT, NPAD, h * HD + nt * 16, kk * 32, lane);
                c = wmma_f16(a, bb, c);
            }
            int ln = lane & 15, mh = lane >> 4;
#pragma unroll
            for (int r = 0; r < 8; ++r)
                outh[(size_t)(mt * 16 + mh * 8 + r) * DIM + h * HD + nt * 16 + ln] = (_Float16)c[r];
        }
        __syncthreads();
    }

    // ---- stage 6: proj GEMM (reuse tmpf as f32 result) ---------------------
    float* projf = tmpf;
    for (int t = wave * 6; t < wave * 6 + 6; ++t) {
        int mt = t / 12, nt = t % 12;
        float bv = b_proj[nt * 16 + (lane & 15)];
        v8f c;
#pragma unroll
        for (int r = 0; r < 8; ++r) c[r] = bv;
#pragma unroll
        for (int kk = 0; kk < 6; ++kk) {
            v16h a  = ldA (outh, DIM, mt * 16, kk * 32, lane);
            v16h bb = ldBT(wproj_h, DIM, nt * 16, kk * 32, lane);
            c = wmma_f16(a, bb, c);
        }
        stC(projf, 200, mt * 16, nt * 16, lane, c);
    }
    __syncthreads();

    // ---- stage 7: LayerNorm1 + residual, scatter with inverse roll ---------
    if (tid < NTOK) {
        const float* y = projf + (size_t)tid * 200;
        float s = 0.f, s2 = 0.f;
        for (int d = 0; d < DIM; ++d) { float v = y[d]; s += v; s2 += v * v; }
        float mean = s / DIM;
        float var  = s2 / DIM - mean * mean;
        stat[tid]      = mean;
        stat[64 + tid] = rsqrtf(var + 1e-5f);
    }
    __syncthreads();
    for (int e = tid; e < NTOK * 48; e += 256) {
        int m = e / 48, g = e % 48;
        int sr = wr * WS_ + m / WS_, sc = wc * WS_ + m % WS_;
        int r = sr + SHIFT_; if (r >= H_) r -= H_;
        int c = sc + SHIFT_; if (c >= W_) c -= W_;
        size_t base = ((size_t)b * LTOK + r * W_ + c) * DIM + g * 4;
        float4 xv = *(const float4*)(x + base);
        float mean = stat[m], rstd = stat[64 + m];
        const float* y = projf + (size_t)m * 200 + g * 4;
        float4 o;
        o.x = xv.x + (y[0] - mean) * rstd * ln1_g[g * 4 + 0] + ln1_b[g * 4 + 0];
        o.y = xv.y + (y[1] - mean) * rstd * ln1_g[g * 4 + 1] + ln1_b[g * 4 + 1];
        o.z = xv.z + (y[2] - mean) * rstd * ln1_g[g * 4 + 2] + ln1_b[g * 4 + 2];
        o.w = xv.w + (y[3] - mean) * rstd * ln1_g[g * 4 + 3] + ln1_b[g * 4 + 3];
        *(float4*)(x1 + base) = o;
    }
}

// ---- fused MLP: x1 -> GELU(fc1) -> fc2 -> LN2 + residual -------------------
// LDS: x1h [0,24576) 64x192 f16 | hb [24576,122880) 64x768 f16
//      yf  [122880,174080) 64x200 f32 | stat [174080,174592)
#define SMEM_MLP 174592

__global__ __launch_bounds__(256) void swin_mlp_kernel(
    const float* __restrict__ x1, const float* __restrict__ fc1_b,
    const float* __restrict__ fc2_b, const float* __restrict__ ln2_g,
    const float* __restrict__ ln2_b, const _Float16* __restrict__ fc1h,
    const _Float16* __restrict__ fc2h, float* __restrict__ out)
{
    extern __shared__ float4 smemv[];
    char* smem = (char*)smemv;
    _Float16* x1h = (_Float16*)(smem);
    _Float16* hb  = (_Float16*)(smem + 24576);
    float*    yf  = (float*)   (smem + 122880);
    float*    stat= (float*)   (smem + 174080);

    const int tid = threadIdx.x, lane = tid & 31, wave = tid >> 5;
    const size_t tok0 = (size_t)blockIdx.x * NPAD;

    for (int e = tid; e < NPAD * 48; e += 256) {
        int m = e / 48, g = e % 48;
        float4 v4 = *(const float4*)(x1 + (tok0 + m) * DIM + g * 4);
        v4h hv;
        hv[0] = (_Float16)v4.x; hv[1] = (_Float16)v4.y;
        hv[2] = (_Float16)v4.z; hv[3] = (_Float16)v4.w;
        *(v4h*)(x1h + (size_t)m * DIM + g * 4) = hv;
    }
    __syncthreads();

    // fc1: (64x192)@(192->768)^T : 4x48 tiles, 24 per wave; exact GELU epilogue
    for (int t = wave * 24; t < wave * 24 + 24; ++t) {
        int mt = t / 48, nt = t % 48;
        float bv = fc1_b[nt * 16 + (lane & 15)];
        v8f c;
#pragma unroll
        for (int r = 0; r < 8; ++r) c[r] = bv;
#pragma unroll
        for (int kk = 0; kk < 6; ++kk) {
            v16h a  = ldA (x1h, DIM, mt * 16, kk * 32, lane);
            v16h bb = ldBT(fc1h, DIM, nt * 16, kk * 32, lane);
            c = wmma_f16(a, bb, c);
        }
        int ln = lane & 15, mh = lane >> 4;
#pragma unroll
        for (int r = 0; r < 8; ++r) {
            float v = c[r];
            float gl = 0.5f * v * (1.f + erff(v * 0.70710678f));
            hb[(size_t)(mt * 16 + mh * 8 + r) * FFN + nt * 16 + ln] = (_Float16)gl;
        }
    }
    __syncthreads();

    // fc2: (64x768)@(768->192)^T : 4x12 tiles, 6 per wave, K=24 steps
    for (int t = wave * 6; t < wave * 6 + 6; ++t) {
        int mt = t / 12, nt = t % 12;
        float bv = fc2_b[nt * 16 + (lane & 15)];
        v8f c;
#pragma unroll
        for (int r = 0; r < 8; ++r) c[r] = bv;
        for (int kk = 0; kk < 24; ++kk) {
            v16h a  = ldA (hb, FFN, mt * 16, kk * 32, lane);
            v16h bb = ldBT(fc2h, FFN, nt * 16, kk * 32, lane);
            c = wmma_f16(a, bb, c);
        }
        stC(yf, 200, mt * 16, nt * 16, lane, c);
    }
    __syncthreads();

    if (tid < NPAD) {
        const float* y = yf + (size_t)tid * 200;
        float s = 0.f, s2 = 0.f;
        for (int d = 0; d < DIM; ++d) { float v = y[d]; s += v; s2 += v * v; }
        float mean = s / DIM;
        float var  = s2 / DIM - mean * mean;
        stat[tid]      = mean;
        stat[64 + tid] = rsqrtf(var + 1e-5f);
    }
    __syncthreads();

    for (int e = tid; e < NPAD * 48; e += 256) {
        int m = e / 48, g = e % 48;
        size_t base = (tok0 + m) * DIM + g * 4;
        float4 xv = *(const float4*)(x1 + base);
        float mean = stat[m], rstd = stat[64 + m];
        const float* y = yf + (size_t)m * 200 + g * 4;
        float4 o;
        o.x = xv.x + (y[0] - mean) * rstd * ln2_g[g * 4 + 0] + ln2_b[g * 4 + 0];
        o.y = xv.y + (y[1] - mean) * rstd * ln2_g[g * 4 + 1] + ln2_b[g * 4 + 1];
        o.z = xv.z + (y[2] - mean) * rstd * ln2_g[g * 4 + 2] + ln2_b[g * 4 + 2];
        o.w = xv.w + (y[3] - mean) * rstd * ln2_g[g * 4 + 3] + ln2_b[g * 4 + 3];
        *(float4*)(out + base) = o;
    }
}

// ---------------------------------------------------------------------------

extern "C" void kernel_launch(void* const* d_in, const int* in_sizes, int n_in,
                              void* d_out, int out_size, void* d_ws, size_t ws_size,
                              hipStream_t stream)
{
    (void)in_sizes; (void)n_in; (void)out_size; (void)ws_size;

    const float* x     = (const float*)d_in[0];
    const float* wqkv  = (const float*)d_in[1];
    const float* qb    = (const float*)d_in[2];
    const float* vb    = (const float*)d_in[3];
    const float* lsc   = (const float*)d_in[4];
    const float* cw1   = (const float*)d_in[5];
    const float* cb1   = (const float*)d_in[6];
    const float* cw2   = (const float*)d_in[7];
    const float* wproj = (const float*)d_in[8];
    const float* bproj = (const float*)d_in[9];
    const float* ln1g  = (const float*)d_in[10];
    const float* ln1b  = (const float*)d_in[11];
    const float* ln2g  = (const float*)d_in[12];
    const float* ln2b  = (const float*)d_in[13];
    const float* f1w   = (const float*)d_in[14];
    const float* f1b   = (const float*)d_in[15];
    const float* f2w   = (const float*)d_in[16];
    const float* f2b   = (const float*)d_in[17];

    char* ws = (char*)d_ws;
    _Float16* wh    = (_Float16*)(ws + WOFF_QKVH);   // contiguous 4-weight f16 block
    _Float16* qkvh  = (_Float16*)(ws + WOFF_QKVH);
    _Float16* projh = (_Float16*)(ws + WOFF_PROJH);
    _Float16* fc1h  = (_Float16*)(ws + WOFF_FC1H);
    _Float16* fc2h  = (_Float16*)(ws + WOFF_FC2H);
    float*    rpb   = (float*)   (ws + WOFF_RPB);
    float*    x1    = (float*)   (ws + WOFF_X1);
    float*    out   = (float*)d_out;

    prep_weights_kernel<<<216, 256, 0, stream>>>(wqkv, wproj, f1w, f2w, wh);
    prep_rpb_kernel<<<1, 256, 0, stream>>>(cw1, cb1, cw2, rpb);
    swin_attn_kernel<<<BATCH * NWIN, 256, SMEM_ATTN, stream>>>(
        x, qb, vb, lsc, bproj, ln1g, ln1b, qkvh, projh, rpb, x1);
    swin_mlp_kernel<<<TOKS / NPAD, 256, SMEM_MLP, stream>>>(
        x1, f1b, f2b, ln2g, ln2b, fc1h, fc2h, out);
}
